// MultiHeadAttention_52578989638094
// MI455X (gfx1250) — matrix operand order repaired and checked by reading, not compiled, not run on hardware
//
#include <hip/hip_runtime.h>
#include <hip/hip_bf16.h>
#include <stdint.h>

// ---------------------------------------------------------------------------
// MI455X (gfx1250) multi-head attention forward, bf16 WMMA + f32 accumulate.
//   B=2, S=2048, D=1024, H=16, DK=64
// Pipeline: 4x weight transpose (f32 -> bf16 W^T), 3x proj GEMM,
//           flash-attn (wave32 WMMA, peeled causal tail), out GEMM (f32 out).
// ---------------------------------------------------------------------------

typedef __attribute__((ext_vector_type(16))) __bf16 v16bf;
typedef __attribute__((ext_vector_type(8)))  __bf16 v8bf;
typedef __attribute__((ext_vector_type(8)))  float  v8f;

static constexpr int BB = 2;
static constexpr int SS = 2048;
static constexpr int DD = 1024;
static constexpr int HH = 16;
static constexpr int DK = 64;
static constexpr int MM = BB * SS;   // 4096 rows

// ---------------------------------------------------------------------------
// Weight transpose + bf16 convert: WT[n][k] = (bf16)W[k][n].
// ---------------------------------------------------------------------------
__global__ __launch_bounds__(256) void wtrans(const float* __restrict__ W,
                                              void* __restrict__ WTv) {
  __bf16* WT = (__bf16*)WTv;
  __shared__ float tile[32][33];
  const int tx = threadIdx.x;   // 0..31
  const int ty = threadIdx.y;   // 0..7
  const int n0 = blockIdx.x * 32;
  const int k0 = blockIdx.y * 32;
#pragma unroll
  for (int j = 0; j < 4; ++j)
    tile[ty + 8 * j][tx] = W[(size_t)(k0 + ty + 8 * j) * DD + n0 + tx];
  __syncthreads();
#pragma unroll
  for (int j = 0; j < 4; ++j)
    WT[(size_t)(n0 + ty + 8 * j) * DD + k0 + tx] = (__bf16)tile[tx][ty + 8 * j];
}

// ---------------------------------------------------------------------------
// GEMM: C[M,N] = scale*(A[M,DD] @ W[DD,N] + bias), W given as bf16 W^T [N,DD].
// MODE 0: A f32, store bf16 head-split [B,H,S,DK]            (q, k)
// MODE 1: A f32, store bf16 head-split transposed [B,H,DK,S] (v^T)
// MODE 2: A bf16 row-major, store f32 row-major [M,N]        (final proj)
// ---------------------------------------------------------------------------
template <int MODE>
__global__ __launch_bounds__(256) void gemm_wmma(const void* __restrict__ Aptr,
                                                 const void* __restrict__ WTv,
                                                 const float* __restrict__ bias,
                                                 void* __restrict__ Cptr,
                                                 float scale) {
  const __bf16* WT = (const __bf16*)WTv;
  const int lane = threadIdx.x & 31;
  const int col  = lane & 15;
  const int hf   = lane >> 4;
  const int wave = threadIdx.x >> 5;
  const int m0 = blockIdx.x * 128 + (wave & 3) * 32;
  const int n0 = blockIdx.y * 64 + (wave >> 2) * 32;

  v8f acc[2][2] = {};

  for (int k0 = 0; k0 < DD; k0 += 32) {
    v16bf a[2];
#pragma unroll
    for (int fr = 0; fr < 2; ++fr) {
      const int row = m0 + fr * 16 + col;
      if (MODE == 2) {
        const __bf16* p = (const __bf16*)Aptr + (size_t)row * DD + k0 + 8 * hf;
        v8bf lo = *(const v8bf*)(p);
        v8bf hi = *(const v8bf*)(p + 16);
#pragma unroll
        for (int i = 0; i < 8; ++i) { a[fr][i] = lo[i]; a[fr][i + 8] = hi[i]; }
      } else {
        const float* p = (const float*)Aptr + (size_t)row * DD + k0 + 8 * hf;
        float4 f0 = *(const float4*)(p);
        float4 f1 = *(const float4*)(p + 4);
        float4 g0 = *(const float4*)(p + 16);
        float4 g1 = *(const float4*)(p + 20);
        a[fr][0] = (__bf16)f0.x;  a[fr][1] = (__bf16)f0.y;
        a[fr][2] = (__bf16)f0.z;  a[fr][3] = (__bf16)f0.w;
        a[fr][4] = (__bf16)f1.x;  a[fr][5] = (__bf16)f1.y;
        a[fr][6] = (__bf16)f1.z;  a[fr][7] = (__bf16)f1.w;
        a[fr][8]  = (__bf16)g0.x; a[fr][9]  = (__bf16)g0.y;
        a[fr][10] = (__bf16)g0.z; a[fr][11] = (__bf16)g0.w;
        a[fr][12] = (__bf16)g1.x; a[fr][13] = (__bf16)g1.y;
        a[fr][14] = (__bf16)g1.z; a[fr][15] = (__bf16)g1.w;
      }
    }
    v16bf b[2];
#pragma unroll
    for (int fc = 0; fc < 2; ++fc) {
      const int n = n0 + fc * 16 + col;
      const __bf16* wp = WT + (size_t)n * DD + k0 + 8 * hf;
      v8bf lo = *(const v8bf*)(wp);
      v8bf hi = *(const v8bf*)(wp + 16);
#pragma unroll
      for (int i = 0; i < 8; ++i) { b[fc][i] = lo[i]; b[fc][i + 8] = hi[i]; }
    }
#pragma unroll
    for (int fr = 0; fr < 2; ++fr)
#pragma unroll
      for (int fc = 0; fc < 2; ++fc)
        acc[fr][fc] = __builtin_amdgcn_wmma_f32_16x16x32_bf16(
            false, a[fr], false, b[fc], (short)0, acc[fr][fc], false, false);
  }

#pragma unroll
  for (int fc = 0; fc < 2; ++fc) {
    const int n = n0 + fc * 16 + col;
    const float bv = bias[n];
#pragma unroll
    for (int fr = 0; fr < 2; ++fr) {
#pragma unroll
      for (int r = 0; r < 8; ++r) {
        const float v = (acc[fr][fc][r] + bv) * scale;
        const int m = m0 + fr * 16 + r + 8 * hf;
        if (MODE == 2) {
          ((float*)Cptr)[(size_t)m * DD + n] = v;
        } else {
          const int bb = m >> 11;       // m / S
          const int s  = m & (SS - 1);
          const int h  = n >> 6;        // n / DK
          const int dk = n & (DK - 1);
          size_t idx;
          if (MODE == 0)
            idx = (((size_t)(bb * HH + h)) * SS + s) * DK + dk;
          else
            idx = (((size_t)(bb * HH + h)) * DK + dk) * SS + s;
          ((__bf16*)Cptr)[idx] = (__bf16)v;
        }
      }
    }
  }
}

// ---------------------------------------------------------------------------
// One 32-key flash-attention chunk. MASKED=true only for the final chunk of a
// query tile (the only one crossing the causal diagonal); mask is branchless.
// 1/sqrt(DK) is pre-folded into q, so raw scores are used directly.
// ---------------------------------------------------------------------------
template <bool MASKED, bool PREFETCH>
__device__ __forceinline__ void attn_chunk(int k0, int q, int hf, int col,
                                           const __bf16* __restrict__ kb,
                                           const __bf16* __restrict__ vb,
                                           const v16bf (&bq)[2], v8f (&o)[4],
                                           float& mstat, float& lstat) {
  if (PREFETCH) {
    // Each active lane prefetches one row of the NEXT chunk (ISA 10.5: every
    // EXEC lane prefetches its address; whole cacheline; no counter traffic).
    __builtin_prefetch(kb + (size_t)(k0 + 32 + (threadIdx.x & 31)) * DK, 0, 3);
    __builtin_prefetch(vb + (size_t)(threadIdx.x & 31) * SS + k0 + 32, 0, 3);
    __builtin_prefetch(vb + (size_t)(32 + (threadIdx.x & 31)) * SS + k0 + 32, 0, 3);
  }
  // ---- scores S^T = K @ Q^T: two 16-key C frags, K-dim = DK in 2 steps -----
  v8f c0 = {}, c1 = {};
#pragma unroll
  for (int c = 0; c < 2; ++c) {
    const __bf16* p = kb + (size_t)(k0 + col) * DK + c * 32 + 8 * hf;
    v8bf lo = *(const v8bf*)(p);
    v8bf hi = *(const v8bf*)(p + 16);
    v16bf ka;
#pragma unroll
    for (int i = 0; i < 8; ++i) { ka[i] = lo[i]; ka[i + 8] = hi[i]; }
    c0 = __builtin_amdgcn_wmma_f32_16x16x32_bf16(false, ka, false, bq[c],
                                                 (short)0, c0, false, false);
    const __bf16* p2 = kb + (size_t)(k0 + 16 + col) * DK + c * 32 + 8 * hf;
    v8bf lo2 = *(const v8bf*)(p2);
    v8bf hi2 = *(const v8bf*)(p2 + 16);
    v16bf ka2;
#pragma unroll
    for (int i = 0; i < 8; ++i) { ka2[i] = lo2[i]; ka2[i + 8] = hi2[i]; }
    c1 = __builtin_amdgcn_wmma_f32_16x16x32_bf16(false, ka2, false, bq[c],
                                                 (short)0, c1, false, false);
  }
  // ---- (optional) branchless causal mask; chunk max ------------------------
  float smax = -1e30f;
#pragma unroll
  for (int r = 0; r < 8; ++r) {
    float x0 = c0[r];
    float x1 = c1[r];
    if (MASKED) {
      x0 = (k0 + r + 8 * hf <= q) ? x0 : -1e30f;
      x1 = (k0 + 16 + r + 8 * hf <= q) ? x1 : -1e30f;
      c0[r] = x0; c1[r] = x1;
    }
    smax = fmaxf(smax, fmaxf(x0, x1));
  }
  smax = fmaxf(smax, __shfl_xor(smax, 16, 32));
  const float mnew = fmaxf(mstat, smax);
  const float rs = __expf(mstat - mnew);
  float p0[8], p1[8];
  float lsum = 0.f;
#pragma unroll
  for (int r = 0; r < 8; ++r) {
    p0[r] = __expf(c0[r] - mnew);
    p1[r] = __expf(c1[r] - mnew);
    lsum += p0[r] + p1[r];
  }
  lsum += __shfl_xor(lsum, 16, 32);
  lstat = lstat * rs + lsum;
  mstat = mnew;
#pragma unroll
  for (int d = 0; d < 4; ++d)
#pragma unroll
    for (int r = 0; r < 8; ++r) o[d][r] *= rs;

  // ---- build P as B operand (K=keys, N=queries): source lanes for lane l
  // are exactly {l, l^16}, so ONE shfl_xor(16) per r suffices:
  //   u = hf ? p0 : p1 ; w = shfl_xor(u,16) gives partner's p0 (h=0 lanes)
  //   or partner's p1 (h=1 lanes).
  v16bf pb;
#pragma unroll
  for (int r = 0; r < 8; ++r) {
    const float u = hf ? p0[r] : p1[r];
    const float w = __shfl_xor(u, 16, 32);
    pb[r]     = (__bf16)(hf ? w : p0[r]);      // key = 16h + r     @ lane col
    pb[r + 8] = (__bf16)(hf ? p1[r] : w);      // key = 16h + 8 + r @ lane col+16
  }
  // ---- O^T += V^T @ P ------------------------------------------------------
#pragma unroll
  for (int d = 0; d < 4; ++d) {
    const __bf16* vp = vb + (size_t)(d * 16 + col) * SS + k0 + 16 * hf;
    v8bf lo = *(const v8bf*)(vp);
    v8bf hi = *(const v8bf*)(vp + 8);
    v16bf va;
#pragma unroll
    for (int i = 0; i < 8; ++i) { va[i] = lo[i]; va[i + 8] = hi[i]; }
    o[d] = __builtin_amdgcn_wmma_f32_16x16x32_bf16(false, va, false, pb,
                                                   (short)0, o[d], false, false);
  }
}

// ---------------------------------------------------------------------------
// Flash attention: one wave per 16-query tile per (b,h).
// ---------------------------------------------------------------------------
__global__ __launch_bounds__(256) void attn_wmma(const void* __restrict__ qhv,
                                                 const void* __restrict__ khv,
                                                 const void* __restrict__ vTv,
                                                 void* __restrict__ aov) {
  const __bf16* qh = (const __bf16*)qhv;
  const __bf16* kh = (const __bf16*)khv;
  const __bf16* vT = (const __bf16*)vTv;
  __bf16* ao = (__bf16*)aov;

  const int lane = threadIdx.x & 31;
  const int col  = lane & 15;
  const int hf   = lane >> 4;

  const int gw = blockIdx.x * 8 + (threadIdx.x >> 5);  // 4096 wave-tiles
  const int bh = gw >> 7;          // / (S/16)
  const int q0 = (gw & 127) << 4;

  const __bf16* qb = qh + (size_t)bh * SS * DK;
  const __bf16* kb = kh + (size_t)bh * SS * DK;
  const __bf16* vb = vT + (size_t)bh * DK * SS;

  // Q^T B-fragments (loop invariant): lane = query col, K(dk) = c*32+8*hf+i (i<8), +16 for i>=8
  v16bf bq[2];
#pragma unroll
  for (int c = 0; c < 2; ++c) {
    const __bf16* p = qb + (size_t)(q0 + col) * DK + c * 32 + 8 * hf;
    v8bf lo = *(const v8bf*)(p);
    v8bf hi = *(const v8bf*)(p + 16);
#pragma unroll
    for (int i = 0; i < 8; ++i) { bq[c][i] = lo[i]; bq[c][i + 8] = hi[i]; }
  }

  v8f o[4] = {};                 // O^T accum: d-chunks 0..3, query = col
  float mstat = -1e30f, lstat = 0.f;
  const int q = q0 + col;
  const int nchunks = (q0 + 16 + 31) >> 5;   // last chunk is the only masked one

  for (int j = 0; j < nchunks - 1; ++j)
    attn_chunk<false, true>(j << 5, q, hf, col, kb, vb, bq, o, mstat, lstat);
  attn_chunk<true, false>((nchunks - 1) << 5, q, hf, col, kb, vb, bq, o,
                          mstat, lstat);

  // ---- normalize + store bf16 row-major [B,S,D]; a lane's 8 elements of one
  // d-chunk are CONSECUTIVE d indices -> pack into one b128 store per chunk.
  const float inv = 1.0f / lstat;
  const int bb = bh >> 4, h = bh & 15;
  __bf16* rowp = ao + ((size_t)(bb * SS + q0 + col)) * DD + h * DK + 8 * hf;
#pragma unroll
  for (int d = 0; d < 4; ++d) {
    v8bf outv;
#pragma unroll
    for (int r = 0; r < 8; ++r) outv[r] = (__bf16)(o[d][r] * inv);
    *(v8bf*)(rowp + d * 16) = outv;
  }
}

// ---------------------------------------------------------------------------
extern "C" void kernel_launch(void* const* d_in, const int* in_sizes, int n_in,
                              void* d_out, int out_size, void* d_ws, size_t ws_size,
                              hipStream_t stream) {
  const float* Q  = (const float*)d_in[0];
  const float* K  = (const float*)d_in[1];
  const float* V  = (const float*)d_in[2];
  const float* Wq = (const float*)d_in[3];
  const float* bq = (const float*)d_in[4];
  const float* Wk = (const float*)d_in[5];
  const float* bk = (const float*)d_in[6];
  const float* Wv = (const float*)d_in[7];
  const float* bv = (const float*)d_in[8];
  const float* Wo = (const float*)d_in[9];
  const float* bo = (const float*)d_in[10];
  // d_in[11] is the causal mask; applied analytically (it is exactly tril).

  char* ws = (char*)d_ws;
  const size_t tsz = (size_t)MM * DD * 2;  // 8 MB per bf16 activation tensor
  const size_t wsz = (size_t)DD * DD * 2;  // 2 MB per bf16 W^T
  void* qh  = (void*)(ws + 0 * tsz);
  void* kh  = (void*)(ws + 1 * tsz);
  void* vT  = (void*)(ws + 2 * tsz);
  void* at  = (void*)(ws + 3 * tsz);
  void* WqT = (void*)(ws + 4 * tsz + 0 * wsz);
  void* WkT = (void*)(ws + 4 * tsz + 1 * wsz);
  void* WvT = (void*)(ws + 4 * tsz + 2 * wsz);
  void* WoT = (void*)(ws + 4 * tsz + 3 * wsz);

  dim3 tb(32, 8);
  dim3 tg(DD / 32, DD / 32);               // 32x32 transpose tiles
  wtrans<<<tg, tb, 0, stream>>>(Wq, WqT);
  wtrans<<<tg, tb, 0, stream>>>(Wk, WkT);
  wtrans<<<tg, tb, 0, stream>>>(Wv, WvT);
  wtrans<<<tg, tb, 0, stream>>>(Wo, WoT);

  const float inv_sqrt_dk = 0.125f;        // 1/sqrt(64), folded into q
  dim3 gg(MM / 128, DD / 64);              // 32 x 16 blocks, 256 threads
  gemm_wmma<0><<<gg, 256, 0, stream>>>(Q, WqT, bq, qh, inv_sqrt_dk);
  gemm_wmma<0><<<gg, 256, 0, stream>>>(K, WkT, bk, kh, 1.0f);
  gemm_wmma<1><<<gg, 256, 0, stream>>>(V, WvT, bv, vT, 1.0f);
  attn_wmma<<<dim3(512), 256, 0, stream>>>(qh, kh, vT, at);
  gemm_wmma<2><<<gg, 256, 0, stream>>>(at, WoT, bo, d_out, 1.0f);
}
